// ExtractorAttn_13005160972328
// MI455X (gfx1250) — matrix operand order repaired and checked by reading, not compiled, not run on hardware
//
#include <hip/hip_runtime.h>
#include <hip/hip_bf16.h>

typedef __attribute__((ext_vector_type(16))) __bf16 v16bf;
typedef __attribute__((ext_vector_type(8)))  float  v8f;
typedef unsigned int u32x4 __attribute__((ext_vector_type(4)));
#define AS3 __attribute__((address_space(3)))

#define CC   64      // channels per input
#define HH   128
#define WW   128
#define HID  128
#define CK   2048    // 2C * K * K reduction length
#define MT   32      // pixels per workgroup (two 16-row WMMA A-tiles)
#define XSTR 2056    // padded LDS row stride (halves) to avoid bank conflicts

__device__ __forceinline__ unsigned short f2bf(float f) {
  unsigned int u = __float_as_uint(f);
  u += 0x7FFFu + ((u >> 16) & 1u);           // round-to-nearest-even
  return (unsigned short)(u >> 16);
}
__device__ __forceinline__ float bf2f(unsigned short h) {
  return __uint_as_float(((unsigned int)h) << 16);
}
__device__ __forceinline__ float leaky(float x) { return x >= 0.f ? x : 0.01f * x; }

union ABPack { struct { u32x4 lo; u32x4 hi; } q; v16bf v; };

// LDS loads through an opaque addrspace(3) base -> ds_load base_vgpr + imm.
__device__ __forceinline__ void loadA(ABPack& A, const AS3 unsigned short* base, int k) {
  A.q.lo = *(const AS3 u32x4*)(base + k);        // K = k .. k+7   (this half-wave)
  A.q.hi = *(const AS3 u32x4*)(base + k + 16);   // K = k+16 .. k+23
}
__device__ __forceinline__ void loadB(ABPack& B, const unsigned short* base, int k) {
  B.q.lo = *(const u32x4*)(base + k);            // K = k .. k+7   (this half-wave)
  B.q.hi = *(const u32x4*)(base + k + 16);       // K = k+16 .. k+23
}

__global__ void pack_w1_bf16(const float* __restrict__ w1,
                             unsigned short* __restrict__ dst, int n) {
  int i = blockIdx.x * blockDim.x + threadIdx.x;
  if (i < n) dst[i] = f2bf(w1[i]);
}

__global__ __launch_bounds__(256)
void extractor_attn_fused(const float* __restrict__ src,
                          const float* __restrict__ tgt,
                          const float* __restrict__ flow,
                          const unsigned short* __restrict__ w1b,
                          const float* __restrict__ b1,
                          const float* __restrict__ w2,
                          const float* __restrict__ b2,
                          float* __restrict__ out) {
  __shared__ unsigned short Xs0[16][XSTR];  // pixels 0..15  x 2048 bf16
  __shared__ unsigned short Xs1[16][XSTR];  // pixels 16..31 x 2048 bf16
  __shared__ float Hs[MT][HID + 4];         // hidden activations
  __shared__ float attnS[MT][16];           // attention weights per pixel

  const int t  = threadIdx.x;
  const int b  = blockIdx.z;
  const int hy = blockIdx.y;
  const int w0 = blockIdx.x * MT;

  // ---------------- Phase A: build x = [block_target ; block_source] in LDS
  #pragma unroll
  for (int j = 0; j < 2; ++j) {
    const int combo = t + j * 256;    // 512 combos = 32 pixels x 16 taps
    const int p   = combo >> 4;       // pixel within tile
    const int pos = combo & 15;       // tap index = ki*4+kj
    const int ki = pos >> 2, kj = pos & 3;
    const int w = w0 + p;
    unsigned short* xr = (p < 16) ? &Xs0[p][0] : &Xs1[p - 16][0];

    // target: integer-shifted window (flow = 0 -> exact sample)
    {
      const int y = hy + ki - 1;
      const int x = w  + kj - 1;
      const bool valid = (y >= 0) && (y < HH) && (x >= 0) && (x < WW);
      const float* tb = tgt + (size_t)b * CC * (HH * WW) + (valid ? (y * WW + x) : 0);
      for (int c = 0; c < CC; ++c) {
        float v = valid ? tb[c * (HH * WW)] : 0.f;
        xr[c * 16 + pos] = f2bf(v);
      }
    }
    // source: bilinear window at flow offset
    {
      const float fx = flow[((size_t)(b * 2 + 0) * HH + hy) * WW + w];
      const float fy = flow[((size_t)(b * 2 + 1) * HH + hy) * WW + w];
      const float ysf = (float)(hy + ki - 1) + fy;
      const float xsf = (float)(w  + kj - 1) + fx;
      const float y0f = floorf(ysf), x0f = floorf(xsf);
      const float wy = ysf - y0f, wx = xsf - x0f;
      const int y0 = (int)y0f, x0 = (int)x0f;
      float wgt[4] = { (1.f - wy) * (1.f - wx), (1.f - wy) * wx,
                       wy * (1.f - wx),         wy * wx };
      int idx[4];
      #pragma unroll
      for (int q = 0; q < 4; ++q) {
        int y = y0 + (q >> 1), x = x0 + (q & 1);
        bool v = (y >= 0) && (y < HH) && (x >= 0) && (x < WW);
        if (!v) wgt[q] = 0.f;
        int yc = min(max(y, 0), HH - 1);
        int xc = min(max(x, 0), WW - 1);
        idx[q] = yc * WW + xc;
      }
      const float* sb = src + (size_t)b * CC * (HH * WW);
      for (int c = 0; c < CC; ++c) {
        const float* scp = sb + c * (HH * WW);
        float v = wgt[0] * scp[idx[0]] + wgt[1] * scp[idx[1]]
                + wgt[2] * scp[idx[2]] + wgt[3] * scp[idx[3]];
        xr[(CC + c) * 16 + pos] = f2bf(v);
      }
    }
  }
  __syncthreads();

  // ---------------- Phase B: h[32x128] = leaky(X[32x2048] * W1^T + b1) via WMMA
  // 8 waves, one 16-out-channel tile each; each B tile feeds TWO WMMAs.
  {
    const int lane  = t & 31;
    const int wv    = t >> 5;
    const int obase = wv * 16;
    const int koffA = (lane < 16) ? 0 : 8;   // ISA 16-bit A layout
    const int koffB = (lane < 16) ? 0 : 8;   // B: same K layout as A
    const int ncol  = lane & 15;

    // Opaque addrspace(3) bases, pinned in VGPRs: stops LLVM from re-folding
    // them into (LDS_base + >64KB constant) recomputed every iteration.
    const AS3 unsigned short* a0base =
        (const AS3 unsigned short*)&Xs0[lane & 15][koffA];   // pixels 0..15
    const AS3 unsigned short* a1base =
        (const AS3 unsigned short*)&Xs1[lane & 15][koffA];   // pixels 16..31
    asm("" : "+v"(a0base));
    asm("" : "+v"(a1base));
    const unsigned short* brow = w1b + (size_t)(obase + ncol) * CK + koffB;

    v8f acc0 = {0.f, 0.f, 0.f, 0.f, 0.f, 0.f, 0.f, 0.f};
    v8f acc1 = {0.f, 0.f, 0.f, 0.f, 0.f, 0.f, 0.f, 0.f};

    ABPack A0a, A1a, Ba, A0b, A1b, Bb;
    loadA(A0a, a0base, 0); loadA(A1a, a1base, 0); loadB(Ba, brow, 0);
    for (int k0 = 0; k0 < CK; k0 += 64) {
      const int k1 = k0 + 32;                 // always < CK
      loadA(A0b, a0base, k1); loadA(A1b, a1base, k1); loadB(Bb, brow, k1);
      acc0 = __builtin_amdgcn_wmma_f32_16x16x32_bf16(
          false, A0a.v, false, Ba.v, (short)0, acc0, false, false);
      acc1 = __builtin_amdgcn_wmma_f32_16x16x32_bf16(
          false, A1a.v, false, Ba.v, (short)0, acc1, false, false);
      const int k2 = (k0 + 64) & (CK - 1);    // branch-free in-bounds wrap
      loadA(A0a, a0base, k2); loadA(A1a, a1base, k2); loadB(Ba, brow, k2);
      acc0 = __builtin_amdgcn_wmma_f32_16x16x32_bf16(
          false, A0b.v, false, Bb.v, (short)0, acc0, false, false);
      acc1 = __builtin_amdgcn_wmma_f32_16x16x32_bf16(
          false, A1b.v, false, Bb.v, (short)0, acc1, false, false);
    }

    // D layout: VGPR r -> M = r (lanes 0-15) / r+8 (lanes 16-31), N = lane%16
    const int mofs = (lane < 16) ? 0 : 8;
    const int o = obase + ncol;
    const float bias = b1[o];
    #pragma unroll
    for (int r = 0; r < 8; ++r) {
      Hs[r + mofs][o]      = leaky(acc0[r] + bias);
      Hs[16 + r + mofs][o] = leaky(acc1[r] + bias);
    }
  }
  __syncthreads();

  // ---------------- Phase C: attn[16] = leaky(W2 * h + b2) per pixel
  #pragma unroll
  for (int j = 0; j < 2; ++j) {
    const int combo = t + j * 256;    // 512 combos = 32 pixels x 16 outputs
    const int p = combo >> 4;
    const int q = combo & 15;
    float s = b2[q];
    const float* w2r = w2 + q * HID;
    for (int o = 0; o < HID; ++o) s += w2r[o] * Hs[p][o];
    attnS[p][q] = leaky(s);
  }
  __syncthreads();

  // ---------------- Phase D: out[c] = (1/16) * sum_pos attn[pos]*block_src[c,pos]
  {
    #pragma unroll
    for (int j = 0; j < 8; ++j) {
      int idx = t + j * 256;       // 2048 outputs = 64 ch x 32 pixels
      int c = idx >> 5;
      int p = idx & 31;
      const unsigned short* xr = (p < 16) ? &Xs0[p][0] : &Xs1[p - 16][0];
      float s = 0.f;
      #pragma unroll
      for (int pos = 0; pos < 16; ++pos)
        s += attnS[p][pos] * bf2f(xr[(CC + c) * 16 + pos]);
      out[(((size_t)b * CC + c) * HH + hy) * WW + (w0 + p)] = s * (1.f / 16.f);
    }
  }
}

extern "C" void kernel_launch(void* const* d_in, const int* in_sizes, int n_in,
                              void* d_out, int out_size, void* d_ws, size_t ws_size,
                              hipStream_t stream) {
  (void)in_sizes; (void)n_in; (void)out_size; (void)ws_size;
  const float* src  = (const float*)d_in[0];
  const float* tgt  = (const float*)d_in[1];
  const float* flow = (const float*)d_in[2];
  const float* w1   = (const float*)d_in[3];
  const float* b1   = (const float*)d_in[4];
  const float* w2   = (const float*)d_in[5];
  const float* b2   = (const float*)d_in[6];
  float* out = (float*)d_out;

  unsigned short* w1b = (unsigned short*)d_ws;   // 128*2048*2 = 512 KB scratch
  const int n = HID * CK;
  pack_w1_bf16<<<(n + 255) / 256, 256, 0, stream>>>(w1, w1b, n);

  dim3 grid(WW / MT, HH, 4);                     // 32-pixel tiles, B=4
  extractor_attn_fused<<<grid, 256, 0, stream>>>(src, tgt, flow, w1b, b1, w2, b2, out);
}
